// SNN_58712202936962
// MI455X (gfx1250) — matrix operand order repaired and checked by reading, not compiled, not run on hardware
//
#include <hip/hip_runtime.h>

#define T_STEPS 512
#define B_DIM   64
#define I_DIM   1024
#define H_DIM   4096
#define O_DIM   1024
#define NBLOCKS 64
#define NTHREADS 256
#define ALPHA 0.9f
#define BETA  0.9f
#define THRESH 1.0f

// Padded LDS row stride for spk1 staging: 1032 elems -> row-to-row bank shift
// of 4 banks, so a 16-lane x 16B ds_load_b128 sweep is conflict-free.
#define SPK1_LDS_STRIDE (I_DIM + 8)

typedef __attribute__((ext_vector_type(16))) __bf16 v16bf;
typedef __attribute__((ext_vector_type(8)))  float  v8f;
typedef int v4i __attribute__((vector_size(16)));  // builtin's pointee type

union FragU { v16bf v; uint4 q[2]; };

#if __has_builtin(__builtin_amdgcn_global_load_async_to_lds_b128) && \
    __has_builtin(__builtin_amdgcn_s_wait_asynccnt)
#define USE_ASYNC_LDS 1
#else
#define USE_ASYNC_LDS 0
#endif

// ---------------------------------------------------------------------------
// bf16 helpers (raw ushort storage so host code never touches __bf16)
// ---------------------------------------------------------------------------
__device__ __forceinline__ unsigned short f32_to_bf16_rne(float f) {
  unsigned u = __builtin_bit_cast(unsigned, f);
  u += 0x7FFFu + ((u >> 16) & 1u);
  return (unsigned short)(u >> 16);
}

// A-fragment (16x32, row major): lane<16 -> K {0..7,16..23}, lane>=16 ->
// K {8..15,24..31}; row = base_row + (lane & 15).
__device__ __forceinline__ v16bf load_fragA(const unsigned short* base, int ld,
                                            int row, int k0, int lane) {
  const unsigned short* p = base + (size_t)row * ld + k0 + ((lane >> 4) << 3);
  FragU f;
  f.q[0] = *(const uint4*)(p);
  f.q[1] = *(const uint4*)(p + 16);
  return f.v;
}

// B-fragment (32x16 of W^T; W row-major [N,K]): lane<16 -> K {0..7,16..23},
// lane>=16 -> K {8..15,24..31}; col = base_col + (lane & 15).
__device__ __forceinline__ v16bf load_fragB(const unsigned short* base, int ld,
                                            int col, int k0, int lane) {
  const unsigned short* p = base + (size_t)col * ld + k0 + ((lane >> 4) << 3);
  FragU f;
  f.q[0] = *(const uint4*)(p);
  f.q[1] = *(const uint4*)(p + 16);
  return f.v;
}

__device__ __forceinline__ v8f wmma_bf16(v16bf a, v16bf b, v8f c) {
  return __builtin_amdgcn_wmma_f32_16x16x32_bf16(
      /*neg_a=*/false, a, /*neg_b=*/false, b,
      /*c_mod=*/(short)0, c, /*reuse_a=*/false, /*reuse_b=*/false);
}

// ---------------------------------------------------------------------------
// Split device-wide barrier: monotonic counters, arrive/wait decoupled so
// independent work overlaps the sync latency.
// ---------------------------------------------------------------------------
__device__ __forceinline__ void bar_arrive(unsigned* cnt) {
  __threadfence();
  __syncthreads();
  if (threadIdx.x == 0)
    __hip_atomic_fetch_add(cnt, 1u, __ATOMIC_RELEASE,
                           __HIP_MEMORY_SCOPE_AGENT);
}

__device__ __forceinline__ void bar_wait(unsigned* cnt, unsigned target) {
  if (threadIdx.x == 0) {
    while (__hip_atomic_load(cnt, __ATOMIC_ACQUIRE,
                             __HIP_MEMORY_SCOPE_AGENT) < target)
      __builtin_amdgcn_s_sleep(2);
  }
  __syncthreads();
}

// Fused synaptic-neuron epilogue over a 16x16 f32 accumulator tile.
__device__ __forceinline__ void neuron_epilogue(float* __restrict__ s,
                                                float* __restrict__ m,
                                                unsigned short* __restrict__ spk,
                                                const v8f acc, int mtile,
                                                int col, int ldn, int lane) {
  const int moff = (lane >> 4) << 3;
#pragma unroll
  for (int j = 0; j < 8; ++j) {
    const int row = mtile * 16 + moff + j;
    const size_t gi = (size_t)row * ldn + col;
    float syn = s[gi], mem = m[gi];
    const float reset = mem > THRESH ? 1.0f : 0.0f;
    syn = ALPHA * syn + acc[j];
    mem = BETA * mem + syn - reset * THRESH;
    s[gi] = syn;
    m[gi] = mem;
    spk[gi] = (mem > THRESH) ? (unsigned short)0x3F80u : (unsigned short)0u;
  }
}

// ---------------------------------------------------------------------------
// Init: zero neuron state, convert weights f32 -> bf16, clear barrier.
// ---------------------------------------------------------------------------
__global__ void snn_init_kernel(const float* __restrict__ w1,
                                const float* __restrict__ w2,
                                unsigned short* __restrict__ w1b,
                                unsigned short* __restrict__ w2b,
                                float* __restrict__ states,
                                unsigned* __restrict__ bar) {
  const size_t g = (size_t)blockIdx.x * blockDim.x + threadIdx.x;
  const size_t stride = (size_t)gridDim.x * blockDim.x;
  const size_t nw1 = (size_t)H_DIM * I_DIM;
  const size_t nw2 = (size_t)O_DIM * H_DIM;
  const size_t nst = 2ull * ((size_t)B_DIM * I_DIM + (size_t)B_DIM * H_DIM +
                             (size_t)B_DIM * O_DIM);
  for (size_t i = g; i < nw1; i += stride) w1b[i] = f32_to_bf16_rne(w1[i]);
  for (size_t i = g; i < nw2; i += stride) w2b[i] = f32_to_bf16_rne(w2[i]);
  for (size_t i = g; i < nst; i += stride) states[i] = 0.0f;
  if (g < 8) bar[g] = 0u;
}

// ---------------------------------------------------------------------------
// Persistent fused SNN kernel.
// Per step: waitA -> [stage spk1->LDS, GEMM1+L2 fuse] -> arriveB ->
//           [L1 neuron for t+1, arriveA]  (overlaps barrier B) ->
//           waitB -> [GEMM2+L3 fuse].
// spk1/spk2 are double-buffered to allow one step of inter-block skew.
// ---------------------------------------------------------------------------
__global__ __launch_bounds__(NTHREADS, 1)
void snn_persistent_kernel(const float* __restrict__ x,
                           const unsigned short* __restrict__ w1b,
                           const unsigned short* __restrict__ w2b,
                           unsigned short* __restrict__ spk1g,
                           unsigned short* __restrict__ spk2g,
                           float* __restrict__ s1, float* __restrict__ m1,
                           float* __restrict__ s2, float* __restrict__ m2,
                           float* __restrict__ s3, float* __restrict__ m3,
                           float* __restrict__ out,
                           unsigned* __restrict__ bar) {
  unsigned* cntA = bar;
  unsigned* cntB = bar + 16;  // separate cachelines-ish

  const int tid = threadIdx.x;
  const int wave = tid >> 5;  // 8 wave32s per block
  const int lane = tid & 31;
  const int blk = blockIdx.x;

  __shared__ __align__(16) unsigned short spk1s[B_DIM * SPK1_LDS_STRIDE];
  __shared__ float red[4][32][8];  // GEMM2 K-split partial sums

  // Phase-0 mapping: 64 blk * 256 thr * 4 elems = B*I = 65536
  const int g0 = (blk * NTHREADS + tid) * 4;

  // Phase-1 mapping: block owns 64 columns of H; wave -> (ntile, 2 mtiles)
  const int p1_ntile = wave >> 1;
  const int p1_mt0 = (wave & 1) * 2;
  const int p1_col = blk * 64 + p1_ntile * 16 + (lane & 15);
  const int p1_ra0 = p1_mt0 * 16 + (lane & 15);
  const int p1_ra1 = p1_ra0 + 16;

  // Phase-2 mapping: block owns 16 columns of O; wave pair splits K
  const int p2_mt = wave >> 1;
  const int p2_khalf = wave & 1;
  const int p2_col = blk * 16 + (lane & 15);
  const int p2_ra = p2_mt * 16 + (lane & 15);
  const int p2_kbeg = p2_khalf * (H_DIM / 2);

  // ---------------- layer-1 neuron update for one timestep ----------------
  auto phase0 = [&](int t) {
    const float* xp = x + (size_t)t * B_DIM * I_DIM + g0;
    if (t + 1 < T_STEPS)
      __builtin_prefetch(xp + (size_t)B_DIM * I_DIM, 0, 1);
    unsigned short* spk1w = spk1g + (size_t)(t & 1) * (B_DIM * I_DIM);
    const float4 xin = *(const float4*)xp;
    float4 sy = *(float4*)(s1 + g0);
    float4 me = *(float4*)(m1 + g0);
    float xv[4] = {xin.x, xin.y, xin.z, xin.w};
    float sv[4] = {sy.x, sy.y, sy.z, sy.w};
    float mv[4] = {me.x, me.y, me.z, me.w};
    union { unsigned short h[4]; uint2 u; } sp;
#pragma unroll
    for (int j = 0; j < 4; ++j) {
      const float reset = mv[j] > THRESH ? 1.0f : 0.0f;
      sv[j] = ALPHA * sv[j] + xv[j];
      mv[j] = BETA * mv[j] + sv[j] - reset * THRESH;
      sp.h[j] = (mv[j] > THRESH) ? (unsigned short)0x3F80u
                                 : (unsigned short)0u;
    }
    *(float4*)(s1 + g0) = make_float4(sv[0], sv[1], sv[2], sv[3]);
    *(float4*)(m1 + g0) = make_float4(mv[0], mv[1], mv[2], mv[3]);
    *(uint2*)(spk1w + g0) = sp.u;
  };

  phase0(0);
  bar_arrive(cntA);

  for (int t = 0; t < T_STEPS; ++t) {
    bar_wait(cntA, (unsigned)(NBLOCKS * (t + 1)));

    // ---- Stage spk1(t) into LDS (async-to-LDS path when available) ----
    {
      const unsigned short* s1gsrc =
          spk1g + (size_t)(t & 1) * (B_DIM * I_DIM);
#if USE_ASYNC_LDS
#pragma unroll 4
      for (int i = 0; i < 32; ++i) {
        const int c = tid + NTHREADS * i;  // 8192 x 16B chunks
        const int row = c >> 7;
        const int off = (c & 127) << 3;
        __builtin_amdgcn_global_load_async_to_lds_b128(
            (v4i*)(s1gsrc + (size_t)row * I_DIM + off),
            (v4i*)(spk1s + row * SPK1_LDS_STRIDE + off), 0, 0);
      }
      __builtin_amdgcn_s_wait_asynccnt(0);
#else
#pragma unroll 4
      for (int i = 0; i < 32; ++i) {
        const int c = tid + NTHREADS * i;
        const int row = c >> 7;
        const int off = (c & 127) << 3;
        *(uint4*)(spk1s + row * SPK1_LDS_STRIDE + off) =
            *(const uint4*)(s1gsrc + (size_t)row * I_DIM + off);
      }
#endif
      __syncthreads();
    }

    // -------- Phase 1: cur2 = spk1 @ w1^T (A from LDS) + layer-2 fuse ------
    {
      unsigned short* spk2w = spk2g + (size_t)(t & 1) * ((size_t)B_DIM * H_DIM);
      v8f acc0 = {};
      v8f acc1 = {};
#pragma unroll 4
      for (int k0 = 0; k0 < I_DIM; k0 += 32) {
        const v16bf bf = load_fragB(w1b, I_DIM, p1_col, k0, lane);
        const v16bf a0 =
            load_fragA(spk1s, SPK1_LDS_STRIDE, p1_ra0, k0, lane);
        const v16bf a1 =
            load_fragA(spk1s, SPK1_LDS_STRIDE, p1_ra1, k0, lane);
        acc0 = wmma_bf16(a0, bf, acc0);
        acc1 = wmma_bf16(a1, bf, acc1);
      }
      neuron_epilogue(s2, m2, spk2w, acc0, p1_mt0, p1_col, H_DIM, lane);
      neuron_epilogue(s2, m2, spk2w, acc1, p1_mt0 + 1, p1_col, H_DIM, lane);
    }
    bar_arrive(cntB);

    // ---- Layer-1 neuron for t+1 overlaps barrier-B latency ----
    if (t + 1 < T_STEPS) {
      phase0(t + 1);
      bar_arrive(cntA);
    }

    bar_wait(cntB, (unsigned)(NBLOCKS * (t + 1)));

    // -------- Phase 2: cur3 = spk2 @ w2^T (WMMA bf16) + layer-3 fuse -------
    {
      const unsigned short* spk2r =
          spk2g + (size_t)(t & 1) * ((size_t)B_DIM * H_DIM);
      v8f acc = {};
#pragma unroll 4
      for (int k0 = p2_kbeg; k0 < p2_kbeg + H_DIM / 2; k0 += 32) {
        const v16bf a = load_fragA(spk2r, H_DIM, p2_ra, k0, lane);
        const v16bf b = load_fragB(w2b, H_DIM, p2_col, k0, lane);
        acc = wmma_bf16(a, b, acc);
      }
      if (p2_khalf) {
#pragma unroll
        for (int j = 0; j < 8; ++j) red[p2_mt][lane][j] = acc[j];
      }
      __syncthreads();
      if (!p2_khalf) {
        const int moff = (lane >> 4) << 3;
#pragma unroll
        for (int j = 0; j < 8; ++j) {
          const float cur = acc[j] + red[p2_mt][lane][j];
          const int row = p2_mt * 16 + moff + j;
          const size_t gi = (size_t)row * O_DIM + p2_col;
          float syn = s3[gi], mem = m3[gi];
          const float reset = mem > THRESH ? 1.0f : 0.0f;
          syn = ALPHA * syn + cur;
          mem = BETA * mem + syn - reset * THRESH;
          s3[gi] = syn;
          m3[gi] = mem;
          out[(size_t)t * B_DIM * O_DIM + gi] = (mem > THRESH) ? 1.0f : 0.0f;
        }
      }
      __syncthreads();  // protect red[] before next step's phase 2
    }
  }
}

// ---------------------------------------------------------------------------
extern "C" void kernel_launch(void* const* d_in, const int* in_sizes, int n_in,
                              void* d_out, int out_size, void* d_ws,
                              size_t ws_size, hipStream_t stream) {
  (void)in_sizes; (void)n_in; (void)out_size; (void)ws_size;
  const float* x = (const float*)d_in[0];   // [T,B,I]
  const float* w1 = (const float*)d_in[1];  // [H,I]
  const float* w2 = (const float*)d_in[2];  // [O,H]
  float* out = (float*)d_out;               // [T,B,O]

  char* p = (char*)d_ws;
  auto carve = [&](size_t bytes) -> char* {
    char* r = p;
    p += (bytes + 255) & ~(size_t)255;
    return r;
  };
  unsigned short* w1b = (unsigned short*)carve(2ull * H_DIM * I_DIM);
  unsigned short* w2b = (unsigned short*)carve(2ull * O_DIM * H_DIM);
  unsigned short* spk1 = (unsigned short*)carve(2ull * 2 * B_DIM * I_DIM);
  unsigned short* spk2 = (unsigned short*)carve(2ull * 2 * B_DIM * H_DIM);
  const size_t nstate = 2ull * ((size_t)B_DIM * I_DIM + (size_t)B_DIM * H_DIM +
                                (size_t)B_DIM * O_DIM);
  float* states = (float*)carve(4ull * nstate);
  unsigned* bar = (unsigned*)carve(256);

  float* s1 = states;
  float* m1 = s1 + (size_t)B_DIM * I_DIM;
  float* s2 = m1 + (size_t)B_DIM * I_DIM;
  float* m2 = s2 + (size_t)B_DIM * H_DIM;
  float* s3 = m2 + (size_t)B_DIM * H_DIM;
  float* m3 = s3 + (size_t)B_DIM * O_DIM;

  snn_init_kernel<<<512, 256, 0, stream>>>(w1, w2, w1b, w2b, states, bar);
  snn_persistent_kernel<<<NBLOCKS, NTHREADS, 0, stream>>>(
      x, w1b, w2b, spk1, spk2, s1, m1, s2, m2, s3, m3, out, bar);
}
